// SignalEmbedder_33801392619793
// MI455X (gfx1250) — compile-verified, matrix operand order checked by reading, not run on hardware
//
#include <hip/hip_runtime.h>
#include <hip/hip_bf16.h>
#include <stdint.h>

#define BSZ     64
#define DLEN    65536
#define N_NODES 1024
#define N_FEATS 256
#define EMBED   256
#define M_ROWS  (BSZ * N_NODES)   // 65536 GEMM rows

typedef __attribute__((ext_vector_type(16))) __bf16 v16bf;
typedef __attribute__((ext_vector_type(8)))  float  v8f;
typedef __attribute__((ext_vector_type(4)))  unsigned int u32x4;
typedef __attribute__((ext_vector_type(8)))  int    i32x8;
typedef __attribute__((ext_vector_type(4)))  int    i32x4;

// ---------------------------------------------------------------------------
// K0: split W (f32, [EMBED][N_FEATS]) into bf16 hi/lo, stored directly in the
// per-lane WMMA B-fragment order for 16x16x32 bf16:
//   packed[((et*8 + kt)*32 + lane)*16 + i]  holds  B[k][e] = W[e][k]
//   with e = et*16 + (lane&15),  k = kt*32 + (lane<16 ? 0 : 16) + i
// ---------------------------------------------------------------------------
__global__ void pack_w_kernel(const float* __restrict__ W,
                              __bf16* __restrict__ whi,
                              __bf16* __restrict__ wlo) {
  int tid  = blockIdx.x * blockDim.x + threadIdx.x;   // 65536 elements
  int i    = tid & 15;
  int lane = (tid >> 4) & 31;
  int kt   = (tid >> 9) & 7;
  int et   = tid >> 12;
  int k = kt * 32 + ((lane < 16) ? 0 : 16) + i;
  int e = et * 16 + (lane & 15);
  float w  = W[e * N_FEATS + k];
  __bf16 h = (__bf16)w;
  __bf16 l = (__bf16)(w - (float)h);
  whi[tid] = h;
  wlo[tid] = l;
}

// ---------------------------------------------------------------------------
// K1: scatter values into dense obs + set per-row feature bitset.
// ---------------------------------------------------------------------------
__global__ void scatter_kernel(const float* __restrict__ vals,
                               const int*   __restrict__ node_ids,
                               const int*   __restrict__ feat_ids,
                               float*        __restrict__ obs,
                               unsigned int* __restrict__ bits) {
  int tid = blockIdx.x * blockDim.x + threadIdx.x;    // BSZ*DLEN = 4.19M
  int b   = tid >> 16;                                 // DLEN == 65536
  int n   = node_ids[tid];
  int f   = feat_ids[tid];
  int row = b * N_NODES + n;
  obs[(size_t)row * N_FEATS + f] = vals[tid];
  atomicOr(&bits[row * 8 + (f >> 5)], 1u << (f & 31));
}

// ---------------------------------------------------------------------------
// K2: popcount bitset -> inv_count per row; also emit missing-node mask.
// ---------------------------------------------------------------------------
__global__ void count_kernel(const unsigned int* __restrict__ bits,
                             float* __restrict__ inv,
                             float* __restrict__ missing) {
  int r = blockIdx.x * blockDim.x + threadIdx.x;      // 65536 rows
  int c = 0;
#pragma unroll
  for (int w = 0; w < 8; ++w) c += __popc(bits[r * 8 + w]);
  inv[r]     = c ? (1.0f / (float)c) : 0.0f;          // nan_to_num(x/0) -> 0
  missing[r] = c ? 0.0f : 1.0f;
}

// ---------------------------------------------------------------------------
// K3: split-bf16 WMMA GEMM with TDM async A-tile fetch.
// One block per 16-row M-tile; 8 waves cover E=256 (wave w -> eBase = w*32).
// Wave 0 DMAs the 16x256 f32 A tile into LDS via tensor_load_to_lds, all 256
// threads convert it once into per-lane bf16 hi/lo fragments in LDS, then
// each wave runs the 8-step K loop (6 wmmas per step) off ds_load_b128.
// ---------------------------------------------------------------------------
__global__ __launch_bounds__(256) void gemm_kernel(
    const float*  __restrict__ obs,
    const float*  __restrict__ inv,
    const __bf16* __restrict__ whi,
    const __bf16* __restrict__ wlo,
    const float*  __restrict__ bias,
    const float*  __restrict__ gemb,
    float*        __restrict__ out) {
  __shared__ __align__(16) float  sA[16 * N_FEATS];      // 16 KB raw f32 tile
  __shared__ __align__(32) __bf16 sHi[8 * 32 * 16];      // 8 KB hi fragments
  __shared__ __align__(32) __bf16 sLo[8 * 32 * 16];      // 8 KB lo fragments

  const int lane    = threadIdx.x & 31;
  const int wave    = threadIdx.x >> 5;
  const int rowBase = blockIdx.x * 16;

  // ---- TDM: DMA the 16x256 f32 A tile into LDS (wave 0 issues; D# uniform) --
  if (threadIdx.x < 32) {
    unsigned long long gaddr =
        (unsigned long long)(uintptr_t)(obs + (size_t)rowBase * N_FEATS);
    unsigned int lds_off = (unsigned int)(uintptr_t)(&sA[0]);

    u32x4 g0;
    g0[0] = 1u;                                          // count=1, user D#
    g0[1] = lds_off;                                     // lds_addr (bytes)
    g0[2] = (unsigned int)(gaddr & 0xffffffffu);         // global_addr[31:0]
    g0[3] = (unsigned int)((gaddr >> 32) & 0x1ffffffu)   // global_addr[56:32]
            | (2u << 30);                                // type = 2 ("image")

    i32x8 g1;
    g1[0] = (int)(2u << 16);      // workgroup_mask=0 (no cluster), data_size=4B
    g1[1] = (int)(256u << 16);    // atomic_barrier_addr=0, tensor_dim0=256 (lo)
    g1[2] = (int)(16u << 16);     // tensor_dim0 hi=0, tensor_dim1=16 (lo)
    g1[3] = (int)(256u << 16);    // tensor_dim1 hi=0, tile_dim0=256
    g1[4] = 16;                   // tile_dim1=16, tile_dim2=0 (2D tile)
    g1[5] = 256;                  // tensor_dim0_stride = 256 (lo32)
    g1[6] = 0;                    // stride hi, tensor_dim1_stride lo = 0
    g1[7] = 0;

    i32x4 gz4 = {0, 0, 0, 0};                     // groups 2/3 unused (<=2D)
    i32x8 gz8 = {0, 0, 0, 0, 0, 0, 0, 0};         // extra group (clang-23 form)
    __builtin_amdgcn_tensor_load_to_lds(g0, g1, gz4, gz4, gz8, 0);
    __builtin_amdgcn_s_wait_tensorcnt(0);
  }
  __syncthreads();

  // ---- Convert f32 tile -> bf16 hi/lo fragments, once per block ------------
  // Fragment layout (16-bit A 16x32, contiguous k): lane<16: m=lane, k elems
  // {0..15}; lane>=16: m=lane-16, k elems {16..31}.
  {
    const int kt   = threadIdx.x >> 5;                  // 0..7
    const int m    = lane & 15;
    const int koff = kt * 32 + ((lane < 16) ? 0 : 16);
    const float s  = inv[rowBase + m];
    const float* src = sA + m * N_FEATS + koff;         // 16 contiguous f32
    __bf16* dh = sHi + (kt * 32 + lane) * 16;
    __bf16* dl = sLo + (kt * 32 + lane) * 16;
#pragma unroll
    for (int i = 0; i < 16; ++i) {
      float v  = src[i] * s;
      __bf16 h = (__bf16)v;
      dh[i] = h;
      dl[i] = (__bf16)(v - (float)h);
    }
  }
  __syncthreads();

  // ---- WMMA K-loop: each wave owns two 16-wide E tiles ---------------------
  const int eBase = wave * 32;
  const int et0   = wave * 2;

  v8f c0 = {};
  v8f c1 = {};

#pragma unroll
  for (int kt = 0; kt < 8; ++kt) {
    const v16bf ah = *(const v16bf*)(sHi + (kt * 32 + lane) * 16);
    const v16bf al = *(const v16bf*)(sLo + (kt * 32 + lane) * 16);
    const v16bf bh0 = *(const v16bf*)(whi + (size_t)((et0 + 0) * 8 + kt) * 512 + lane * 16);
    const v16bf bl0 = *(const v16bf*)(wlo + (size_t)((et0 + 0) * 8 + kt) * 512 + lane * 16);
    const v16bf bh1 = *(const v16bf*)(whi + (size_t)((et0 + 1) * 8 + kt) * 512 + lane * 16);
    const v16bf bl1 = *(const v16bf*)(wlo + (size_t)((et0 + 1) * 8 + kt) * 512 + lane * 16);

    // (a_hi + a_lo)(b_hi + b_lo) ~= hi*hi + hi*lo + lo*hi, f32 accumulate
    c0 = __builtin_amdgcn_wmma_f32_16x16x32_bf16(false, ah, false, bh0, (short)0, c0, false, false);
    c0 = __builtin_amdgcn_wmma_f32_16x16x32_bf16(false, ah, false, bl0, (short)0, c0, false, false);
    c0 = __builtin_amdgcn_wmma_f32_16x16x32_bf16(false, al, false, bh0, (short)0, c0, false, false);
    c1 = __builtin_amdgcn_wmma_f32_16x16x32_bf16(false, ah, false, bh1, (short)0, c1, false, false);
    c1 = __builtin_amdgcn_wmma_f32_16x16x32_bf16(false, ah, false, bl1, (short)0, c1, false, false);
    c1 = __builtin_amdgcn_wmma_f32_16x16x32_bf16(false, al, false, bh1, (short)0, c1, false, false);
  }

  // Epilogue: C/D layout: lanes 0-15 -> N=lane, M=j; lanes 16-31 -> N=lane-16, M=8+j
  const int nlane  = lane & 15;
  const int mShift = (lane < 16) ? 0 : 8;
#pragma unroll
  for (int j = 0; j < 8; ++j) {
    int r    = rowBase + mShift + j;
    int node = r & (N_NODES - 1);
    int e0   = eBase + nlane;
    int e1   = eBase + 16 + nlane;
    out[(size_t)r * EMBED + e0] = c0[j] + bias[e0] + gemb[(size_t)node * EMBED + e0];
    out[(size_t)r * EMBED + e1] = c1[j] + bias[e1] + gemb[(size_t)node * EMBED + e1];
  }
}

// ---------------------------------------------------------------------------
// Workspace layout (bytes):
//   [0,              67,108,864)  obs dense  f32  (64*1024*256)
//   [67,108,864,     69,206,016)  bitset     u32  (65536 * 8)
//   [69,206,016,     69,468,160)  inv_count  f32  (65536)
//   [69,468,160,     69,599,232)  W_hi bf16  (65536)
//   [69,599,232,     69,730,304)  W_lo bf16  (65536)
// ---------------------------------------------------------------------------
extern "C" void kernel_launch(void* const* d_in, const int* in_sizes, int n_in,
                              void* d_out, int out_size, void* d_ws, size_t ws_size,
                              hipStream_t stream) {
  const float* flat_inputs = (const float*)d_in[0];
  const int*   node_ids    = (const int*)d_in[1];
  const int*   feat_ids    = (const int*)d_in[2];
  const float* W           = (const float*)d_in[3];
  const float* bias        = (const float*)d_in[4];
  const float* gemb        = (const float*)d_in[5];

  char* ws = (char*)d_ws;
  float*        obs  = (float*)(ws);
  unsigned int* bits = (unsigned int*)(ws + 67108864);
  float*        inv  = (float*)(ws + 69206016);
  __bf16*       whi  = (__bf16*)(ws + 69468160);
  __bf16*       wlo  = (__bf16*)(ws + 69599232);

  float* embedded = (float*)d_out;                        // 64*1024*256
  float* missing  = embedded + (size_t)M_ROWS * EMBED;    // 64*1024

  // Zero dense obs + bitset (fresh every call; graph-capture safe).
  (void)hipMemsetAsync(d_ws, 0, 69206016, stream);

  pack_w_kernel<<<dim3(65536 / 256), dim3(256), 0, stream>>>(W, whi, wlo);

  scatter_kernel<<<dim3((BSZ * DLEN) / 256), dim3(256), 0, stream>>>(
      flat_inputs, node_ids, feat_ids, obs, bits);

  count_kernel<<<dim3(M_ROWS / 256), dim3(256), 0, stream>>>(bits, inv, missing);

  // One block per 16-row M-tile; 8 waves cover the full E=256.
  gemm_kernel<<<dim3(M_ROWS / 16), dim3(256), 0, stream>>>(
      obs, inv, whi, wlo, bias, gemb, embedded);
}
